// PatternEncoder_23441931501599
// MI455X (gfx1250) — matrix operand order repaired and checked by reading, not compiled, not run on hardware
//
#include <hip/hip_runtime.h>
#include <hip/hip_bf16.h>

typedef __attribute__((ext_vector_type(16))) _Float16 v16h;
typedef __attribute__((ext_vector_type(8)))  float    v8f;

// ---------------------------------------------------------------------------
// CDNA5 async copy helpers (gfx1250): global -> LDS without VGPR round-trip,
// tracked by ASYNCcnt (CDNA5 ISA).
// ---------------------------------------------------------------------------
__device__ __forceinline__ void async_ld_lds_b128(void* lds_ptr, const void* gptr)
{
    unsigned           loff = (unsigned)(uintptr_t)lds_ptr;   // LDS aperture: low 32 bits
    unsigned long long ga   = (unsigned long long)(uintptr_t)gptr;
    asm volatile("global_load_async_to_lds_b128 %0, %1, off"
                 :: "v"(loff), "v"(ga) : "memory");
}

__device__ __forceinline__ void wait_async0()
{
    asm volatile("s_wait_asynccnt 0x0" ::: "memory");
}

// ---------------------------------------------------------------------------
// WMMA GEMM:  C[M x Nc] = act(A[M x K] @ W[K x Nc] (+ bias))
//   * block = 256 threads (8 waves), tile = 16(M) x 128(N)
//   * per K-step(32): A tile (16x32 fp32) filled via global_load_async_to_lds,
//     B tile (32x128) staged coalesced, converted to packed f16 pairs and
//     transposed into the per-lane WMMA-B layout in LDS.
//   * one v_wmma_f32_16x16x32_f16 per wave per K-step; f32 accumulate.
// Requires M%16==0, K%32==0, Nc%128==0 (all shapes here satisfy this).
// ---------------------------------------------------------------------------
#define SA_STRIDE 36   // fp32 elems per A row (32 + pad, 16B aligned, bank-spread)
#define SB_STRIDE 20   // packed u32 per B "row" (16 + pad, 16B aligned)

__global__ __launch_bounds__(256)
void gemm_wmma_f16(const float* __restrict__ A, const float* __restrict__ W,
                   const float* __restrict__ bias, float* __restrict__ C,
                   int M, int K, int Nc, int relu)
{
    __shared__ __align__(16) float    sA[16 * SA_STRIDE];        // fp32 A tile
    __shared__ __align__(16) unsigned sB[128 * SB_STRIDE];       // packed f16 B tile

    const int t    = threadIdx.x;
    const int wave = t >> 5;              // 0..7  -> N sub-tile
    const int lane = t & 31;
    const int half = lane >> 4;           // 0 / 1
    const int l16  = lane & 15;

    const int m0 = blockIdx.y * 16;
    const int n0 = blockIdx.x * 128;
    const int kb = half * 8;              // A K sub-base per lane half

    v8f acc = {};

    for (int k0 = 0; k0 < K; k0 += 32) {
        // ---- stage A (16x32 fp32) via async copy: 128 x 16B chunks ----
        if (t < 128) {
            const int row = t >> 3;                   // 0..15
            const int cc  = t & 7;                    // 4-float chunk
            async_ld_lds_b128(&sA[row * SA_STRIDE + cc * 4],
                              A + (size_t)(m0 + row) * K + k0 + cc * 4);
        }
        // ---- stage B (32x128) -> packed f16, transposed to [n][kpair] ----
        #pragma unroll
        for (int it = 0; it < 8; ++it) {
            const int p  = t + it * 256;              // 0..2047
            const int n  = p & 127;
            const int kp = p >> 7;                    // 0..15 (K pair)
            const float w0 = W[(size_t)(k0 + 2 * kp)     * Nc + n0 + n];
            const float w1 = W[(size_t)(k0 + 2 * kp + 1) * Nc + n0 + n];
            union { _Float16 h[2]; unsigned u; } pk;
            pk.h[0] = (_Float16)w0;
            pk.h[1] = (_Float16)w1;
            sB[n * SB_STRIDE + kp] = pk.u;
        }
        wait_async0();
        __syncthreads();

        // ---- per-lane A fragment (ISA 7.12.2 16-bit A 16x32 layout) ----
        const float4 a0 = *(const float4*)&sA[l16 * SA_STRIDE + kb];
        const float4 a1 = *(const float4*)&sA[l16 * SA_STRIDE + kb + 4];
        const float4 a2 = *(const float4*)&sA[l16 * SA_STRIDE + kb + 16];
        const float4 a3 = *(const float4*)&sA[l16 * SA_STRIDE + kb + 20];
        v16h a;
        a[0]  = (_Float16)a0.x; a[1]  = (_Float16)a0.y;
        a[2]  = (_Float16)a0.z; a[3]  = (_Float16)a0.w;
        a[4]  = (_Float16)a1.x; a[5]  = (_Float16)a1.y;
        a[6]  = (_Float16)a1.z; a[7]  = (_Float16)a1.w;
        a[8]  = (_Float16)a2.x; a[9]  = (_Float16)a2.y;
        a[10] = (_Float16)a2.z; a[11] = (_Float16)a2.w;
        a[12] = (_Float16)a3.x; a[13] = (_Float16)a3.y;
        a[14] = (_Float16)a3.z; a[15] = (_Float16)a3.w;

        // ---- per-lane B fragment: packed pairs k=8h..8h+7 and 16+8h..16+8h+7 from LDS ----
        const int nl = wave * 16 + l16;               // local n (0..127)
        const uint4 q0 = *(const uint4*)&sB[nl * SB_STRIDE + half * 4];
        const uint4 q1 = *(const uint4*)&sB[nl * SB_STRIDE + half * 4 + 8];
        union { unsigned u[8]; v16h v; } bu;
        bu.u[0] = q0.x; bu.u[1] = q0.y; bu.u[2] = q0.z; bu.u[3] = q0.w;
        bu.u[4] = q1.x; bu.u[5] = q1.y; bu.u[6] = q1.z; bu.u[7] = q1.w;

        acc = __builtin_amdgcn_wmma_f32_16x16x32_f16(
                  false, a, false, bu.v, (short)0, acc, false, false);

        __syncthreads();   // protect LDS before next stage
    }

    // ---- epilogue: C/D layout N = lane%16, VGPR r -> M = r + 8*(lane/16) ----
    const int n = n0 + wave * 16 + l16;
    const float bv = bias ? bias[n] : 0.0f;
    #pragma unroll
    for (int r = 0; r < 8; ++r) {
        const int m = m0 + r + 8 * half;
        float v = acc[r] + bv;
        if (relu) v = fmaxf(v, 0.0f);
        C[(size_t)m * Nc + n] = v;
    }
}

// ---------------------------------------------------------------------------
// Elementwise / graph kernels
// ---------------------------------------------------------------------------
__global__ void fill_kernel(float* __restrict__ p, float v, int n)
{
    int i = blockIdx.x * blockDim.x + threadIdx.x;
    if (i < n) p[i] = v;
}

__global__ void deg_accum_kernel(const int* __restrict__ col,
                                 float* __restrict__ deg, int E)
{
    int e = blockIdx.x * blockDim.x + threadIdx.x;
    if (e < E) atomicAdd(&deg[col[e]], 1.0f);
}

__global__ void rsqrt_inplace_kernel(float* __restrict__ d, int n)
{
    int i = blockIdx.x * blockDim.x + threadIdx.x;
    if (i < n) {
        float v = d[i];
        d[i] = (v > 0.0f) ? rsqrtf(v) : 0.0f;
    }
}

__global__ void edge_norm_kernel(const int* __restrict__ row,
                                 const int* __restrict__ col,
                                 const float* __restrict__ dinv,
                                 float* __restrict__ norm, int E)
{
    int e = blockIdx.x * blockDim.x + threadIdx.x;
    if (e < E) norm[e] = dinv[row[e]] * dinv[col[e]];
}

// out[col[e], :] += h[row[e], :] * norm[e]   (float4 chunks, atomic scatter)
__global__ __launch_bounds__(256)
void scatter_add_kernel(const float* __restrict__ h, const int* __restrict__ row,
                        const int* __restrict__ col, const float* __restrict__ norm,
                        float* __restrict__ out, int E, int F)
{
    const int tpe  = F >> 2;                  // threads per edge (float4 chunks)
    const int epb  = blockDim.x / tpe;        // edges per block
    const int eloc = threadIdx.x / tpe;
    const int f4   = threadIdx.x % tpe;
    const long long e = (long long)blockIdx.x * epb + eloc;
    if (e >= E) return;

    const int r = row[e];
    const int c = col[e];
    const float nv = norm[e];
    const float4 hv = *(const float4*)(h + (size_t)r * F + f4 * 4);
    float* dst = out + (size_t)c * F + f4 * 4;
    atomicAdd(dst + 0, hv.x * nv);
    atomicAdd(dst + 1, hv.y * nv);
    atomicAdd(dst + 2, hv.z * nv);
    atomicAdd(dst + 3, hv.w * nv);
}

// out = relu(out + h * dinv[i]^2 + bias[f])   (self-loop term fused in)
__global__ void selfloop_bias_relu_kernel(const float* __restrict__ h,
                                          const float* __restrict__ dinv,
                                          const float* __restrict__ bias,
                                          float* __restrict__ out,
                                          int N, int F)
{
    long long idx = (long long)blockIdx.x * blockDim.x + threadIdx.x;
    long long total = (long long)N * F;
    if (idx >= total) return;
    const int i = (int)(idx / F);
    const int f = (int)(idx % F);
    const float di = dinv[i];
    float v = out[idx] + h[idx] * di * di + bias[f];
    out[idx] = fmaxf(v, 0.0f);
}

__global__ void pool_cnt_kernel(const int* __restrict__ batch,
                                float* __restrict__ cnt, int N)
{
    int i = blockIdx.x * blockDim.x + threadIdx.x;
    if (i < N) atomicAdd(&cnt[batch[i]], 1.0f);
}

__global__ void pool_sum_kernel(const float* __restrict__ h,
                                const int* __restrict__ batch,
                                float* __restrict__ gsum, int N, int F)
{
    long long idx = (long long)blockIdx.x * blockDim.x + threadIdx.x;
    long long total = (long long)N * F;
    if (idx >= total) return;
    const int i = (int)(idx / F);
    const int f = (int)(idx % F);
    atomicAdd(&gsum[(size_t)batch[i] * F + f], h[idx]);
}

__global__ void pool_div_kernel(const float* __restrict__ gsum,
                                const float* __restrict__ cnt,
                                float* __restrict__ g, int G, int F)
{
    int idx = blockIdx.x * blockDim.x + threadIdx.x;
    if (idx >= G * F) return;
    const int gi = idx / F;
    g[idx] = gsum[idx] / fmaxf(cnt[gi], 1.0f);
}

// ---------------------------------------------------------------------------
// Host-side orchestration
// ---------------------------------------------------------------------------
static inline void launch_gemm(const float* A, const float* W, const float* bias,
                               float* C, int M, int K, int Nc, int relu,
                               hipStream_t s)
{
    dim3 grid(Nc / 128, M / 16);
    gemm_wmma_f16<<<grid, 256, 0, s>>>(A, W, bias, C, M, K, Nc, relu);
}

extern "C" void kernel_launch(void* const* d_in, const int* in_sizes, int n_in,
                              void* d_out, int out_size, void* d_ws, size_t ws_size,
                              hipStream_t stream)
{
    const float* x     = (const float*)d_in[0];
    const int*   ei    = (const int*)  d_in[1];
    const int*   batch = (const int*)  d_in[2];
    const float* W1    = (const float*)d_in[3];
    const float* b1    = (const float*)d_in[4];
    const float* W2    = (const float*)d_in[5];
    const float* b2    = (const float*)d_in[6];
    const float* Wl1   = (const float*)d_in[7];
    const float* bl1   = (const float*)d_in[8];
    const float* Wl2   = (const float*)d_in[9];
    const float* bl2   = (const float*)d_in[10];
    const float* Wl22  = (const float*)d_in[11];
    const float* bl22  = (const float*)d_in[12];
    const float* Wl23  = (const float*)d_in[13];
    const float* bl23  = (const float*)d_in[14];
    const float* Wl3   = (const float*)d_in[15];
    const float* bl3   = (const float*)d_in[16];

    const int D  = 128;
    const int D2 = 256;
    const int G  = 256;
    const int H  = 1024;
    const int N  = in_sizes[0] / D;
    const int E  = in_sizes[1] / 2;
    const int* row = ei;
    const int* col = ei + E;

    // ---- workspace layout (floats) ----
    float* ws = (float*)d_ws;
    size_t off = 0;
    float* bufA = ws + off; off += (size_t)N * D2;   // h1 pre-agg; later h2 / agg2
    float* bufB = ws + off; off += (size_t)N * D2;   // agg1 -> h1_act
    float* dinv = ws + off; off += (size_t)N;        // deg -> dinv
    float* norm = ws + off; off += (size_t)E;
    float* gsum = ws + off; off += (size_t)G * D;
    float* cnt  = ws + off; off += (size_t)G;
    float* g0   = ws + off; off += (size_t)G * D;
    float* t1   = ws + off; off += (size_t)G * D;
    float* t2   = ws + off; off += (size_t)G * H;
    float* t3   = ws + off; off += (size_t)G * H;
    (void)ws_size; (void)n_in;

    float* h2   = bufA;                        // N x 128 (layer-2 GEMM result)
    float* agg2 = bufA + (size_t)N * D;        // N x 128 (layer-2 aggregate)

    const int TB = 256;

    // ---- degree / norm precompute ----
    fill_kernel<<<(N + TB - 1) / TB, TB, 0, stream>>>(dinv, 1.0f, N); // self-loop
    deg_accum_kernel<<<(E + TB - 1) / TB, TB, 0, stream>>>(col, dinv, E);
    rsqrt_inplace_kernel<<<(N + TB - 1) / TB, TB, 0, stream>>>(dinv, N);
    edge_norm_kernel<<<(E + TB - 1) / TB, TB, 0, stream>>>(row, col, dinv, norm, E);

    // ---- GCN layer 1:  h = x @ W1 ; agg1 = scatter ; relu(agg1 + self + b1) ----
    launch_gemm(x, W1, nullptr, bufA, N, D, D2, 0, stream);
    hipMemsetAsync(bufB, 0, (size_t)N * D2 * sizeof(float), stream);
    {
        const int epb = TB / (D2 / 4);   // 4 edges / block @ F=256
        scatter_add_kernel<<<(E + epb - 1) / epb, TB, 0, stream>>>(
            bufA, row, col, norm, bufB, E, D2);
    }
    {
        long long total = (long long)N * D2;
        selfloop_bias_relu_kernel<<<(int)((total + TB - 1) / TB), TB, 0, stream>>>(
            bufA, dinv, b1, bufB, N, D2);
    }

    // ---- GCN layer 2:  h2 = h1_act @ W2 ; agg2 = scatter ; relu(+self+b2) ----
    launch_gemm(bufB, W2, nullptr, h2, N, D2, D, 0, stream);
    hipMemsetAsync(agg2, 0, (size_t)N * D * sizeof(float), stream);
    {
        const int epb = TB / (D / 4);    // 8 edges / block @ F=128
        scatter_add_kernel<<<(E + epb - 1) / epb, TB, 0, stream>>>(
            h2, row, col, norm, agg2, E, D);
    }
    {
        long long total = (long long)N * D;
        selfloop_bias_relu_kernel<<<(int)((total + TB - 1) / TB), TB, 0, stream>>>(
            h2, dinv, b2, agg2, N, D);
    }

    // ---- global mean pool ----
    hipMemsetAsync(gsum, 0, (size_t)G * D * sizeof(float), stream);
    hipMemsetAsync(cnt,  0, (size_t)G * sizeof(float), stream);
    pool_cnt_kernel<<<(N + TB - 1) / TB, TB, 0, stream>>>(batch, cnt, N);
    {
        long long total = (long long)N * D;
        pool_sum_kernel<<<(int)((total + TB - 1) / TB), TB, 0, stream>>>(
            agg2, batch, gsum, N, D);
    }
    pool_div_kernel<<<(G * D + TB - 1) / TB, TB, 0, stream>>>(gsum, cnt, g0, G, D);

    // ---- MLP head (all WMMA, fused bias+ReLU; final layer -> d_out, no ReLU) ----
    launch_gemm(g0, Wl1,  bl1,  t1, G, D, D, 1, stream);   // 256x128 @ 128x128
    launch_gemm(t1, Wl2,  bl2,  t2, G, D, H, 1, stream);   // -> 256x1024
    launch_gemm(t2, Wl22, bl22, t3, G, H, H, 1, stream);   // 256x1024
    launch_gemm(t3, Wl23, bl23, t2, G, H, H, 1, stream);   // 256x1024 (reuse t2)
    launch_gemm(t2, Wl3,  bl3,  (float*)d_out, G, H, D, 0, stream); // 256x128
    (void)out_size;
}
